// CouplingGNN_26465588478542
// MI455X (gfx1250) — matrix operand order repaired and checked by reading, not compiled, not run on hardware
//
#include <hip/hip_runtime.h>

// ============================================================================
// 2-layer GAT (GATConv w/ edge features) for MI455X (gfx1250, wave32, WMMA).
//
//  * edge-feature projection folded to a 16xH matrix (e never materialized)
//  * node tensors (~20MB) stay L2-resident (192MB L2): edge gather/scatter
//    and f32 atomics run at L2 bandwidth, not HBM
//  * dense GEMMs on v_wmma_f32_16x16x32_f16; LDS tiles stored PRE-SWIZZLED in
//    fragment-ready layout so fragment loads are ds_load_b128 (not u16 storms)
//  * segment softmax: encoded-uint atomicMax + f32 atomic sums
// ============================================================================

typedef __attribute__((ext_vector_type(16))) _Float16 v16h;
typedef __attribute__((ext_vector_type(8)))  _Float16 v8h;
typedef __attribute__((ext_vector_type(8)))  float    v8f;

#define TPB 256

// ---------------------------------------------------------------------------
// WMMA GEMM: C[MxNc] = A[MxK] @ B[KxNc], fp32 row-major global, f16 compute /
// f32 accumulate. 256 threads = 8 waves; 64x64 tile; wave owns a 16x32 strip.
// K%32==0, Nc%64==0, M%16==0 (ragged M zero-padded + guarded stores).
//
// LDS tiles are stored fragment-ready: Asf[mblk][lane][slot] where
//   A element (r,c):  lane = (r&15) + 16*((c>>3)&1)
//                     slot = 2*(((c&7)>>1) + 4*(c>>4)) + (c&1)
//   B element (k,c):  blk = c>>4, lane = (c&15) + 16*((k>>3)&1), slot = 8*(k>>4) + (k&7)
// so a lane's v16h fragment is 16 contiguous f16 -> two ds_load_b128.
// Staging: any aligned 8-run of A-cols / B-rows lands in one lane's 8
// contiguous slots -> one ds_store_b128 per thread per tile.
// ---------------------------------------------------------------------------
__global__ __launch_bounds__(256) void gat_gemm_wmma(const float* __restrict__ A,
                                                     const float* __restrict__ B,
                                                     float* __restrict__ C,
                                                     int M, int K, int Nc) {
  constexpr int FP = 8;  // pad slots: stride 48B keeps 16B align, spreads banks
  __shared__ __align__(32) _Float16 Asf[4][32][16 + FP];
  __shared__ __align__(32) _Float16 Bsf[4][32][16 + FP];

  const int tid  = threadIdx.x;
  const int lane = tid & 31;
  const int wave = tid >> 5;
  const int wm   = wave & 3;   // 16-row block of tile
  const int wn   = wave >> 2;  // 32-col block of tile
  const int row0 = blockIdx.y * 64;
  const int col0 = blockIdx.x * 64;

  // A staging map: tid -> row ar(0..63), col group ac0 in {0,8,16,24}
  const int ar     = tid >> 2;
  const int ac0    = (tid & 3) << 3;
  const int a_blk  = ar >> 4;
  const int a_lane = (ar & 15) + ((ac0 & 8) << 1);
  const int a_slot = (ac0 >> 4) << 3;
  // B staging map: tid -> col bc(0..63), k group bk in {0,8,16,24}
  const int bc     = tid & 63;
  const int bk     = (tid >> 6) << 3;
  const int b_blk  = bc >> 4;
  const int b_lane = (bc & 15) + ((bk & 8) << 1);
  const int b_slot = (bk >> 4) << 3;

  v8f acc0 = {}, acc1 = {};

  for (int k0 = 0; k0 < K; k0 += 32) {
    // ---- stage A: two b128 global reads -> convert -> one b128 LDS store ---
    {
      const int gr = row0 + ar;
      float4 f0 = make_float4(0.f, 0.f, 0.f, 0.f);
      float4 f1 = f0;
      if (gr < M) {
        const float4* src = reinterpret_cast<const float4*>(A + (long)gr * K + k0 + ac0);
        f0 = src[0];
        f1 = src[1];
      }
      v8h pk;
      pk[0] = (_Float16)f0.x; pk[1] = (_Float16)f0.y;
      pk[2] = (_Float16)f0.z; pk[3] = (_Float16)f0.w;
      pk[4] = (_Float16)f1.x; pk[5] = (_Float16)f1.y;
      pk[6] = (_Float16)f1.z; pk[7] = (_Float16)f1.w;
      *reinterpret_cast<v8h*>(&Asf[a_blk][a_lane][a_slot]) = pk;
    }
    // ---- stage B: 8 wave-coalesced reads -> one b128 LDS store ----
    {
      const float* src = B + (long)(k0 + bk) * Nc + col0 + bc;
      v8h pk;
#pragma unroll
      for (int j = 0; j < 8; ++j) pk[j] = (_Float16)src[(long)j * Nc];
      *reinterpret_cast<v8h*>(&Bsf[b_blk][b_lane][b_slot]) = pk;
    }
    __syncthreads();

    // fragment loads: two ds_load_b128 each, register-only assembly
    v8h alo = *reinterpret_cast<const v8h*>(&Asf[wm][lane][0]);
    v8h ahi = *reinterpret_cast<const v8h*>(&Asf[wm][lane][8]);
    v16h afrag = __builtin_shufflevector(alo, ahi, 0, 1, 2, 3, 4, 5, 6, 7,
                                         8, 9, 10, 11, 12, 13, 14, 15);
#pragma unroll
    for (int nf = 0; nf < 2; ++nf) {
      v8h blo = *reinterpret_cast<const v8h*>(&Bsf[wn * 2 + nf][lane][0]);
      v8h bhi = *reinterpret_cast<const v8h*>(&Bsf[wn * 2 + nf][lane][8]);
      v16h bfrag = __builtin_shufflevector(blo, bhi, 0, 1, 2, 3, 4, 5, 6, 7,
                                           8, 9, 10, 11, 12, 13, 14, 15);
      if (nf == 0)
        acc0 = __builtin_amdgcn_wmma_f32_16x16x32_f16(false, afrag, false, bfrag,
                                                      (short)0, acc0, false, false);
      else
        acc1 = __builtin_amdgcn_wmma_f32_16x16x32_f16(false, afrag, false, bfrag,
                                                      (short)0, acc1, false, false);
    }
    __syncthreads();
  }

  // C/D layout: lane -> n = lane&15, m = (lane>>4)*8 + vgpr
  const int n  = lane & 15;
  const int mb = (lane >> 4) * 8;
#pragma unroll
  for (int v = 0; v < 8; ++v) {
    int gm = row0 + wm * 16 + mb + v;
    if (gm < M) {
      C[(long)gm * Nc + col0 + wn * 32 + n]      = acc0[v];
      C[(long)gm * Nc + col0 + wn * 32 + 16 + n] = acc1[v];
    }
  }
}

// ---------------------------------------------------------------------------
// helpers
// ---------------------------------------------------------------------------
__device__ inline unsigned enc_f32(float f) {  // monotone float->uint
  unsigned b = __float_as_uint(f);
  return (b & 0x80000000u) ? ~b : (b | 0x80000000u);
}
__device__ inline float dec_f32(unsigned e) {
  unsigned b = (e & 0x80000000u) ? (e & 0x7fffffffu) : ~e;
  return __uint_as_float(b);
}

// mean(edge_attr) over rows: per-thread 16 accumulators, then 16 atomics
__global__ void gat_mean_accum(const float* __restrict__ ea, float* meansum, int E) {
  int tid = blockIdx.x * blockDim.x + threadIdx.x;
  int nth = gridDim.x * blockDim.x;
  float acc[16];
#pragma unroll
  for (int k = 0; k < 16; ++k) acc[k] = 0.f;
  for (int r = tid; r < E; r += nth) {
    const float* row = ea + (long)r * 16;
#pragma unroll
    for (int k = 0; k < 16; ++k) acc[k] += row[k];
  }
#pragma unroll
  for (int k = 0; k < 16; ++k) atomicAdd(&meansum[k], acc[k]);
}
__global__ void gat_mean_scale(float* meansum, float invE) {
  if (threadIdx.x < 16) meansum[threadIdx.x] *= invE;
}

// fold lin_edge[16,H*C] with att_edge[H,C] -> M[16,H]
__global__ void gat_fold_edge_att(const float* __restrict__ lin_e,
                                  const float* __restrict__ att_e,
                                  float* Mout, int H, int C) {
  int i = blockIdx.x * blockDim.x + threadIdx.x;  // k*H + h
  if (i >= 16 * H) return;
  int k = i / H, hh = i % H;
  float s = 0.f;
  for (int c = 0; c < C; ++c) s += lin_e[k * H * C + hh * C + c] * att_e[hh * C + c];
  Mout[i] = s;
}

// per-node attention scalars: a_s[n,h] = h[n,h,:].att_s[h,:], same for a_d
__global__ void gat_node_att(const float* __restrict__ h,
                             const float* __restrict__ att_s,
                             const float* __restrict__ att_d,
                             float* a_s, float* a_d, int N, int H, int C) {
  int idx = blockIdx.x * blockDim.x + threadIdx.x;
  if (idx >= N * H) return;
  int n = idx / H, hh = idx % H;
  const float* row = h + (long)n * H * C + hh * C;
  const float* vs = att_s + hh * C;
  const float* vd = att_d + hh * C;
  float s = 0.f, d = 0.f;
  for (int c = 0; c < C; ++c) { s += row[c] * vs[c]; d += row[c] * vd[c]; }
  a_s[idx] = s;
  a_d[idx] = d;
}

// pass 1 over edges (incl. self-loops): alpha = leaky_relu(a_s+a_d+a_e),
// store alpha, segment-max via encoded atomicMax (amax_enc pre-zeroed)
__global__ void gat_edge_alpha_max(const int* __restrict__ ei, const float* __restrict__ ea,
                                   const float* __restrict__ mean_ea,
                                   const float* __restrict__ Mfold,
                                   const float* __restrict__ a_s, const float* __restrict__ a_d,
                                   float* __restrict__ alpha, unsigned* __restrict__ amax_enc,
                                   int E, int N, int H) {
  int idx = blockIdx.x * blockDim.x + threadIdx.x;
  int Et = E + N;
  if (idx >= Et) return;
  int s, d;
  const float* row;
  if (idx < E) { s = ei[idx]; d = ei[E + idx]; row = ea + (long)idx * 16; }
  else         { s = d = idx - E; row = mean_ea; }
  float aeh[4] = {0.f, 0.f, 0.f, 0.f};
  for (int k = 0; k < 16; ++k) {
    float rv = row[k];
    for (int hh = 0; hh < H; ++hh) aeh[hh] += rv * Mfold[k * H + hh];
  }
  for (int hh = 0; hh < H; ++hh) {
    float v = a_s[(long)s * H + hh] + a_d[(long)d * H + hh] + aeh[hh];
    v = v > 0.f ? v : 0.2f * v;  // leaky_relu(0.2)
    alpha[(long)idx * H + hh] = v;
    atomicMax(&amax_enc[(long)d * H + hh], enc_f32(v));
  }
}

// pass 2: ex = exp(alpha - amax[dst]) (stored in place), denom[dst] += ex
__global__ void gat_edge_expsum(const int* __restrict__ ei, float* __restrict__ alpha,
                                const unsigned* __restrict__ amax_enc,
                                float* __restrict__ denom, int E, int N, int H) {
  int idx = blockIdx.x * blockDim.x + threadIdx.x;
  int Et = E + N;
  if (idx >= Et) return;
  int d = (idx < E) ? ei[E + idx] : (idx - E);
  for (int hh = 0; hh < H; ++hh) {
    float am = dec_f32(amax_enc[(long)d * H + hh]);
    float ex = __expf(alpha[(long)idx * H + hh] - am);
    alpha[(long)idx * H + hh] = ex;
    atomicAdd(&denom[(long)d * H + hh], ex);
  }
}

// pass 3: agg[dst] += h[src] * (ex/denom[dst]); one wave per edge, lanes on
// channels; all traffic L2-resident (node tables << 192MB L2)
__global__ __launch_bounds__(256) void gat_edge_aggregate(const int* __restrict__ ei,
    const float* __restrict__ hsrc, const float* __restrict__ exw,
    const float* __restrict__ denom, float* __restrict__ agg,
    int E, int N, int H, int C) {
  int Et = E + N;
  int lane = threadIdx.x & 31;
  int wid = (blockIdx.x * blockDim.x + threadIdx.x) >> 5;
  int nwaves = (gridDim.x * blockDim.x) >> 5;
  int HC = H * C;
  for (int e = wid; e < Et; e += nwaves) {
    int s, d;
    if (e < E) { s = ei[e]; d = ei[E + e]; } else { s = d = e - E; }
    float wv[4];
    for (int hh = 0; hh < H; ++hh)
      wv[hh] = exw[(long)e * H + hh] / denom[(long)d * H + hh];
    for (int c = lane; c < HC; c += 32)
      atomicAdd(&agg[(long)d * HC + c], hsrc[(long)s * HC + c] * wv[c / C]);
  }
}

// act = elu(agg + bias), in place
__global__ void gat_bias_elu(float* __restrict__ x, const float* __restrict__ bias,
                             long total, int HC) {
  long i = blockIdx.x * (long)blockDim.x + threadIdx.x;
  if (i >= total) return;
  float v = x[i] + bias[i % HC];
  x[i] = v > 0.f ? v : (__expf(v) - 1.0f);
}

// global mean pool: atomic sums + counts, then divide
__global__ void gat_pool_sum(const float* __restrict__ act, const int* __restrict__ batch,
                             float* sums, float* cnt, int N, int C) {
  long i = blockIdx.x * (long)blockDim.x + threadIdx.x;
  if (i >= (long)N * C) return;
  int n = (int)(i / C), c = (int)(i % C);
  int g = batch[n];
  atomicAdd(&sums[(long)g * C + c], act[i]);
  if (c == 0) atomicAdd(&cnt[g], 1.0f);
}
__global__ void gat_pool_div(const float* __restrict__ sums, const float* __restrict__ cnt,
                             float* __restrict__ out, int total, int C) {
  int i = blockIdx.x * blockDim.x + threadIdx.x;
  if (i >= total) return;
  out[i] = sums[i] / fmaxf(cnt[i / C], 1.0f);
}

// ---------------------------------------------------------------------------
extern "C" void kernel_launch(void* const* d_in, const int* in_sizes, int n_in,
                              void* d_out, int out_size, void* d_ws, size_t ws_size,
                              hipStream_t stream) {
  (void)n_in; (void)out_size; (void)ws_size;
  const float* x      = (const float*)d_in[0];
  const int*   ei     = (const int*)d_in[1];    // [2,E] int32 (JAX x64 off)
  const float* ea     = (const float*)d_in[2];
  const int*   batch  = (const int*)d_in[3];
  const float* W1     = (const float*)d_in[4];
  const float* att_s1 = (const float*)d_in[5];
  const float* att_d1 = (const float*)d_in[6];
  const float* lin_e1 = (const float*)d_in[7];
  const float* att_e1 = (const float*)d_in[8];
  const float* bias1  = (const float*)d_in[9];
  const float* W2     = (const float*)d_in[10];
  const float* att_s2 = (const float*)d_in[11];
  const float* att_d2 = (const float*)d_in[12];
  const float* lin_e2 = (const float*)d_in[13];
  const float* att_e2 = (const float*)d_in[14];
  const float* bias2  = (const float*)d_in[15];

  const int NODE_IN = 128, EDGE_IN = 16, HIDDEN = 64, HEADS = 4, OUT_DIM = 64, G = 64;
  const int N   = in_sizes[0] / NODE_IN;
  const int E   = in_sizes[1] / 2;
  const int Et  = E + N;
  const int HC1 = HEADS * HIDDEN;  // 256

  // ---- workspace carve (zero-init region first, single async memset) ----
  char* p = (char*)d_ws;
  auto carve = [&](size_t bytes) -> char* {
    char* r = p; p += (bytes + 255) & ~(size_t)255; return r;
  };
  char*     z0      = p;
  float*    agg1    = (float*)carve((size_t)N * HC1 * 4);      // -> act1 in place
  float*    agg2    = (float*)carve((size_t)N * OUT_DIM * 4);  // -> act2 in place
  unsigned* amax1   = (unsigned*)carve((size_t)N * HEADS * 4);
  float*    denom1  = (float*)carve((size_t)N * HEADS * 4);
  unsigned* amax2   = (unsigned*)carve((size_t)N * 4);
  float*    denom2  = (float*)carve((size_t)N * 4);
  float*    meansum = (float*)carve(EDGE_IN * 4);
  float*    sums    = (float*)carve((size_t)G * OUT_DIM * 4);
  float*    cnt     = (float*)carve((size_t)G * 4);
  size_t    zbytes  = (size_t)(p - z0);
  float*    h1      = (float*)carve((size_t)N * HC1 * 4);
  float*    h2      = (float*)carve((size_t)N * OUT_DIM * 4);
  float*    as1     = (float*)carve((size_t)N * HEADS * 4);
  float*    ad1     = (float*)carve((size_t)N * HEADS * 4);
  float*    as2     = (float*)carve((size_t)N * 4);
  float*    ad2     = (float*)carve((size_t)N * 4);
  float*    alpha1  = (float*)carve((size_t)Et * HEADS * 4);
  float*    alpha2  = (float*)carve((size_t)Et * 4);
  float*    M1      = (float*)carve(EDGE_IN * HEADS * 4);
  float*    M2      = (float*)carve(EDGE_IN * 4);

  hipMemsetAsync(z0, 0, zbytes, stream);  // capture-safe

  // tiny precompute: mean edge_attr, folded edge-attention matrices
  gat_mean_accum<<<64, 256, 0, stream>>>(ea, meansum, E);
  gat_mean_scale<<<1, 32, 0, stream>>>(meansum, 1.0f / (float)E);
  gat_fold_edge_att<<<1, 64, 0, stream>>>(lin_e1, att_e1, M1, HEADS, HIDDEN);
  gat_fold_edge_att<<<1, 16, 0, stream>>>(lin_e2, att_e2, M2, 1, OUT_DIM);

  const int ETB = (Et + 255) / 256;

  // ---- layer 1 ----
  {
    dim3 grid(HC1 / 64, (N + 63) / 64);
    gat_gemm_wmma<<<grid, 256, 0, stream>>>(x, W1, h1, N, NODE_IN, HC1);
  }
  gat_node_att<<<(N * HEADS + 255) / 256, 256, 0, stream>>>(h1, att_s1, att_d1, as1, ad1, N, HEADS, HIDDEN);
  gat_edge_alpha_max<<<ETB, 256, 0, stream>>>(ei, ea, meansum, M1, as1, ad1, alpha1, amax1, E, N, HEADS);
  gat_edge_expsum<<<ETB, 256, 0, stream>>>(ei, alpha1, amax1, denom1, E, N, HEADS);
  gat_edge_aggregate<<<4096, 256, 0, stream>>>(ei, h1, alpha1, denom1, agg1, E, N, HEADS, HIDDEN);
  gat_bias_elu<<<(int)(((long)N * HC1 + 255) / 256), 256, 0, stream>>>(agg1, bias1, (long)N * HC1, HC1);

  // ---- layer 2 (heads=1, concat=false degenerates to identity) ----
  {
    dim3 grid(OUT_DIM / 64, (N + 63) / 64);
    gat_gemm_wmma<<<grid, 256, 0, stream>>>(agg1, W2, h2, N, HC1, OUT_DIM);
  }
  gat_node_att<<<(N + 255) / 256, 256, 0, stream>>>(h2, att_s2, att_d2, as2, ad2, N, 1, OUT_DIM);
  gat_edge_alpha_max<<<ETB, 256, 0, stream>>>(ei, ea, meansum, M2, as2, ad2, alpha2, amax2, E, N, 1);
  gat_edge_expsum<<<ETB, 256, 0, stream>>>(ei, alpha2, amax2, denom2, E, N, 1);
  gat_edge_aggregate<<<4096, 256, 0, stream>>>(ei, h2, alpha2, denom2, agg2, E, N, 1, OUT_DIM);
  gat_bias_elu<<<(int)(((long)N * OUT_DIM + 255) / 256), 256, 0, stream>>>(agg2, bias2, (long)N * OUT_DIM, OUT_DIM);

  // ---- global mean pool ----
  gat_pool_sum<<<(int)(((long)N * OUT_DIM + 255) / 256), 256, 0, stream>>>(agg2, batch, sums, cnt, N, OUT_DIM);
  gat_pool_div<<<(G * OUT_DIM + 255) / 256, 256, 0, stream>>>(sums, cnt, (float*)d_out, G * OUT_DIM, OUT_DIM);
}
